// Attention_2877628088942
// MI455X (gfx1250) — matrix operand order repaired and checked by reading, not compiled, not run on hardware
//
#include <hip/hip_runtime.h>

// ---------------------------------------------------------------------------
// Sigmoid attention for MI455X (gfx1250), bf16 WMMA pipeline + async LDS copy.
// B=4, N=2048, C=1024, H=16, D=64, M=B*N=8192.
// d_out = [ out: 8192*1024 f32 | attn: 64*2048*2048 f32 ]
// d_ws  = [ qh bf16 8388608 | kh bf16 8388608 | vT bf16 8388608 | ctx f32 8388608 ]
// ---------------------------------------------------------------------------

typedef __bf16 bf16_t;
typedef __attribute__((ext_vector_type(16))) __bf16 v16bf;
typedef __attribute__((ext_vector_type(8)))  float  v8f;

union FragU {
  v16bf v;
  uint4 q[2];
};

union Pack4 {
  bf16_t h[4];
  unsigned long long u;
};

__device__ __forceinline__ v16bf ldfrag(const bf16_t* p0, const bf16_t* p1) {
  FragU f;
  f.q[0] = *(const uint4*)p0;
  f.q[1] = *(const uint4*)p1;
  return f.v;
}

__device__ __forceinline__ v8f wmma_bf16(v16bf a, v16bf b, v8f c) {
  // D = A(16x32 bf16) * B(32x16 bf16) + C(16x16 f32)
  return __builtin_amdgcn_wmma_f32_16x16x32_bf16(false, a, false, b, (short)0, c,
                                                 false, false);
}

// sigmoid(x/8) with cheap transcendentals: v_tanh_f32 + v_fma_f32.
__device__ __forceinline__ float fast_sigmoid_scaled(float x) {
#if __has_builtin(__builtin_amdgcn_tanhf)
  return __builtin_fmaf(0.5f, __builtin_amdgcn_tanhf(x * 0.0625f), 0.5f);
#else
  float e = __builtin_amdgcn_exp2f(x * (-0.125f * 1.44269504088896340736f));
  return __builtin_amdgcn_rcpf(1.0f + e);
#endif
}

// 32-bit LDS offset (addrspace(3) view) of a generic shared-memory pointer.
typedef __attribute__((address_space(3))) const char* lds_cptr_t;
__device__ __forceinline__ unsigned lds_off(const void* p) {
  return (unsigned)(unsigned long long)(lds_cptr_t)p;
}

// Async global->LDS copy of 16 bytes per lane (tracked by ASYNCcnt).
__device__ __forceinline__ void async_copy_b128(unsigned lds, const void* g) {
  asm volatile("global_load_async_to_lds_b128 %0, %1, off"
               :
               : "v"(lds), "v"((unsigned long long)(size_t)g)
               : "memory");
}

__device__ __forceinline__ void wait_async0() {
  asm volatile("s_wait_asynccnt 0x0" ::: "memory");
}

// ---------------------------------------------------------------------------
// Y[8192,1024] = X[8192,1024] @ W[1024,1024]^T   (torch Linear, y = x W^T)
// MODE 0: write bf16 head-split  [B,H,N,D]   (q, k)
// MODE 1: write bf16 head-split-T [B,H,D,N]  (v, pre-transposed for attn@v)
// MODE 2: write f32 [M,C] + bias             (final projection)
// Block: 128 threads (4 waves), 64x64 output tile, 32-deep k-steps via LDS.
// ---------------------------------------------------------------------------
template <int MODE>
__global__ __launch_bounds__(128) void gemm_qkv(
    const float* __restrict__ X, const float* __restrict__ W,
    const float* __restrict__ bias, bf16_t* __restrict__ obf,
    float* __restrict__ of) {
  constexpr int K = 1024;
  __shared__ alignas(16) bf16_t At[64 * 40];  // 64 rows, 40 bf16 stride (80 B)
  __shared__ alignas(16) bf16_t Bt[64 * 40];

  const int t = threadIdx.x;
  const int lane = t & 31;
  const int w = t >> 5;
  const int l15 = lane & 15, hi = lane >> 4;
  const int m0 = blockIdx.y * 64, n0 = blockIdx.x * 64;
  const int wm = (w >> 1) * 32, wn = (w & 1) * 32;

  v8f acc[2][2] = {};

  for (int k0 = 0; k0 < K; k0 += 32) {
    // Stage 64x32 f32 tiles of X and W, converting to bf16 in LDS.
#pragma unroll
    for (int i = 0; i < 4; ++i) {
      int cid = t + i * 128;            // 0..511 chunks of 4 floats
      int row = cid >> 3, kc = cid & 7;
      float4 xa = *(const float4*)(X + (size_t)(m0 + row) * K + k0 + kc * 4);
      float4 xb = *(const float4*)(W + (size_t)(n0 + row) * K + k0 + kc * 4);
      Pack4 pa, pb;
      pa.h[0] = (bf16_t)xa.x; pa.h[1] = (bf16_t)xa.y;
      pa.h[2] = (bf16_t)xa.z; pa.h[3] = (bf16_t)xa.w;
      pb.h[0] = (bf16_t)xb.x; pb.h[1] = (bf16_t)xb.y;
      pb.h[2] = (bf16_t)xb.z; pb.h[3] = (bf16_t)xb.w;
      *(unsigned long long*)(At + row * 40 + kc * 4) = pa.u;
      *(unsigned long long*)(Bt + row * 40 + kc * 4) = pb.u;
    }
    // Prefetch next k-step's X/W tile rows while this tile is consumed.
    if (k0 + 32 < K) {
      int row = t >> 3, kc = t & 7;
      __builtin_prefetch(X + (size_t)(m0 + row) * K + (k0 + 32) + kc * 4, 0, 0);
      __builtin_prefetch(W + (size_t)(n0 + row) * K + (k0 + 32) + kc * 4, 0, 0);
    }
    __syncthreads();

    v16bf af[2], bfr[2];
#pragma unroll
    for (int mt = 0; mt < 2; ++mt) {
      const bf16_t* b = At + (wm + mt * 16 + l15) * 40;
      af[mt] = ldfrag(b + hi * 8, b + 16 + hi * 8);   // A: K=8*hi.., K=16+8*hi..
    }
#pragma unroll
    for (int nt = 0; nt < 2; ++nt) {
      const bf16_t* b = Bt + (wn + nt * 16 + l15) * 40;  // B: K=8*hi.., K=16+8*hi..
      bfr[nt] = ldfrag(b + hi * 8, b + 16 + hi * 8);
    }
#pragma unroll
    for (int mt = 0; mt < 2; ++mt)
#pragma unroll
      for (int nt = 0; nt < 2; ++nt)
        acc[mt][nt] = wmma_bf16(af[mt], bfr[nt], acc[mt][nt]);
    __syncthreads();
  }

  // Epilogue: C/D layout -> element (M = r + 8*hi, N = l15) per VGPR r.
#pragma unroll
  for (int mt = 0; mt < 2; ++mt)
#pragma unroll
    for (int nt = 0; nt < 2; ++nt)
#pragma unroll
      for (int r = 0; r < 8; ++r) {
        int mm = m0 + wm + mt * 16 + r + 8 * hi;
        int cc = n0 + wn + nt * 16 + l15;
        float val = acc[mt][nt][r];
        if constexpr (MODE == 2) {
          __builtin_nontemporal_store(val + bias[cc],
                                      of + (size_t)mm * 1024 + cc);
        } else {
          int b = mm >> 11, ns = mm & 2047;   // batch, seq
          int h = cc >> 6,  d = cc & 63;      // head, head-dim
          if constexpr (MODE == 0)
            obf[((size_t)(b * 16 + h) * 2048 + ns) * 64 + d] = (bf16_t)val;
          else
            obf[((size_t)(b * 16 + h) * 64 + d) * 2048 + ns] = (bf16_t)val;
        }
      }
}

// ---------------------------------------------------------------------------
// Per (b,h): S = sigmoid(q k^T / 8)  [written to attn, f32 NT-stores],
//            O = S @ v               (accumulated, written to ctx f32 [M,C]).
// Block: 128 threads (4 waves), 64 q-rows per block, j-loop over 2048 in 64s.
// k/v tiles are double-buffered in LDS via async global->LDS copies
// (ASYNCcnt), overlapping the copy of tile j+1 with compute on tile j.
// q fragments stay in registers for the whole j-loop.
// ---------------------------------------------------------------------------
__global__ __launch_bounds__(128) void attn_sigmoid(
    const bf16_t* __restrict__ qh, const bf16_t* __restrict__ kh,
    const bf16_t* __restrict__ vT, float* __restrict__ attn,
    float* __restrict__ ctx) {
  __shared__ alignas(16) bf16_t Sl[64 * 72];      // 64x64 S tile (bf16)
  __shared__ alignas(16) bf16_t Kt[2][64 * 72];   // k tile [j][d], dbl-buffered
  __shared__ alignas(16) bf16_t Vt[2][64 * 72];   // v tile [d][j], dbl-buffered

  const int t = threadIdx.x, lane = t & 31, w = t >> 5;
  const int l15 = lane & 15, hi = lane >> 4;
  const int wm = (w >> 1) * 32, wn = (w & 1) * 32;
  const int bh = blockIdx.y;
  const int i0 = blockIdx.x * 64;

  const bf16_t* qp = qh + (size_t)bh * (2048 * 64);
  const bf16_t* kp = kh + (size_t)bh * (2048 * 64);
  const bf16_t* vp = vT + (size_t)bh * (64 * 2048);
  float* ap = attn + (size_t)bh * 2048 * 2048;

  // Issue async copies of the 64x64 k and v tiles for column block jt.
  // 512 16B chunks per tile; 128 threads x 4 chunks each.
  auto issue_tiles = [&](int jt, int bb) {
#pragma unroll
    for (int i = 0; i < 4; ++i) {
      int cid = t + i * 128;
      int row = cid >> 3, col = cid & 7;      // row: 64, col: 8 chunks of 8 bf16
      async_copy_b128(lds_off(&Kt[bb][row * 72 + col * 8]),
                      kp + (size_t)(jt + row) * 64 + col * 8);
      async_copy_b128(lds_off(&Vt[bb][row * 72 + col * 8]),
                      vp + (size_t)row * 2048 + jt + col * 8);
    }
  };

  // q fragments (rows i0+wm .. +32, full D=64), kept in registers.
  v16bf qf[2][2];
#pragma unroll
  for (int mt = 0; mt < 2; ++mt)
#pragma unroll
    for (int kc = 0; kc < 2; ++kc) {
      const bf16_t* b = qp + (size_t)(i0 + wm + mt * 16 + l15) * 64 + kc * 32;
      qf[mt][kc] = ldfrag(b + hi * 8, b + 16 + hi * 8);
    }

  issue_tiles(0, 0);
  int cur = 0;

  v8f oacc[2][2] = {};

  for (int j0 = 0; j0 < 2048; j0 += 64) {
    // Tiles[cur] ready; Sl from previous iteration fully consumed.
    wait_async0();
    __syncthreads();

    // Overlap: start copying the next j-tile into the other buffer.
    if (j0 + 64 < 2048) issue_tiles(j0 + 64, cur ^ 1);

    // ---- WMMA1: S tile (wave: 32x32 of the block's 64x64) ----
    v16bf kf[2][2];
#pragma unroll
    for (int nt = 0; nt < 2; ++nt)
#pragma unroll
      for (int kc = 0; kc < 2; ++kc) {
        const bf16_t* b =
            Kt[cur] + (wn + nt * 16 + l15) * 72 + kc * 32;
        kf[nt][kc] = ldfrag(b + hi * 8, b + 16 + hi * 8);
      }
    v8f sacc[2][2] = {};
#pragma unroll
    for (int mt = 0; mt < 2; ++mt)
#pragma unroll
      for (int nt = 0; nt < 2; ++nt) {
        sacc[mt][nt] = wmma_bf16(qf[mt][0], kf[nt][0], sacc[mt][nt]);
        sacc[mt][nt] = wmma_bf16(qf[mt][1], kf[nt][1], sacc[mt][nt]);
      }

    // ---- sigmoid; stream attn to HBM; stage bf16 S tile in LDS ----
#pragma unroll
    for (int mt = 0; mt < 2; ++mt)
#pragma unroll
      for (int nt = 0; nt < 2; ++nt)
#pragma unroll
        for (int r = 0; r < 8; ++r) {
          int il = wm + mt * 16 + r + 8 * hi;
          int jl = wn + nt * 16 + l15;
          float s = fast_sigmoid_scaled(sacc[mt][nt][r]);
          __builtin_nontemporal_store(
              s, ap + (size_t)(i0 + il) * 2048 + (j0 + jl));
          Sl[il * 72 + jl] = (bf16_t)s;
        }
    __syncthreads();

    // ---- WMMA2: O (wave: rows wm..+32, d = wn..+32) += S @ v ----
    v16bf sf[2][2], vf[2][2];
#pragma unroll
    for (int mt = 0; mt < 2; ++mt)
#pragma unroll
      for (int kc = 0; kc < 2; ++kc) {
        const bf16_t* b = Sl + (wm + mt * 16 + l15) * 72 + kc * 32;
        sf[mt][kc] = ldfrag(b + hi * 8, b + 16 + hi * 8);
      }
#pragma unroll
    for (int nt = 0; nt < 2; ++nt)
#pragma unroll
      for (int kc = 0; kc < 2; ++kc) {
        const bf16_t* b =
            Vt[cur] + (wn + nt * 16 + l15) * 72 + kc * 32;
        vf[nt][kc] = ldfrag(b + hi * 8, b + 16 + hi * 8);
      }
#pragma unroll
    for (int mt = 0; mt < 2; ++mt)
#pragma unroll
      for (int nt = 0; nt < 2; ++nt) {
        oacc[mt][nt] = wmma_bf16(sf[mt][0], vf[nt][0], oacc[mt][nt]);
        oacc[mt][nt] = wmma_bf16(sf[mt][1], vf[nt][1], oacc[mt][nt]);
      }
    cur ^= 1;
  }

  // ---- write context (merged heads): ctx[b*2048+i, h*64+d] ----
  int b_ = bh >> 4, h_ = bh & 15;
#pragma unroll
  for (int mt = 0; mt < 2; ++mt)
#pragma unroll
    for (int nt = 0; nt < 2; ++nt)
#pragma unroll
      for (int r = 0; r < 8; ++r) {
        int i = i0 + wm + mt * 16 + r + 8 * hi;
        int c = h_ * 64 + wn + nt * 16 + l15;
        ctx[(size_t)(b_ * 2048 + i) * 1024 + c] = oacc[mt][nt][r];
      }
}

// ---------------------------------------------------------------------------
extern "C" void kernel_launch(void* const* d_in, const int* in_sizes, int n_in,
                              void* d_out, int out_size, void* d_ws,
                              size_t ws_size, hipStream_t stream) {
  (void)in_sizes; (void)n_in; (void)out_size; (void)ws_size;

  const float* x_q = (const float*)d_in[0];
  const float* x_k = (const float*)d_in[1];
  const float* x_v = (const float*)d_in[2];
  const float* Wq  = (const float*)d_in[3];
  const float* Wk  = (const float*)d_in[4];
  const float* Wv  = (const float*)d_in[5];
  const float* Wp  = (const float*)d_in[6];
  const float* bp  = (const float*)d_in[7];

  const size_t HE = 8388608;  // 4*16*2048*64 elements per q/k/v tensor
  float* out  = (float*)d_out;
  float* attn = out + HE;     // out is [8192,1024] f32 = 8388608

  bf16_t* qh = (bf16_t*)d_ws;
  bf16_t* kh = qh + HE;
  bf16_t* vT = kh + HE;
  float* ctx = (float*)((char*)d_ws + 3 * HE * sizeof(bf16_t));

  dim3 gg(1024 / 64, 8192 / 64);  // (16, 128)
  gemm_qkv<0><<<gg, 128, 0, stream>>>(x_q, Wq, nullptr, qh, nullptr);
  gemm_qkv<0><<<gg, 128, 0, stream>>>(x_k, Wk, nullptr, kh, nullptr);
  gemm_qkv<1><<<gg, 128, 0, stream>>>(x_v, Wv, nullptr, vT, nullptr);

  attn_sigmoid<<<dim3(2048 / 64, 64), 128, 0, stream>>>(qh, kh, vT, attn, ctx);

  gemm_qkv<2><<<gg, 128, 0, stream>>>(ctx, Wp, bp, nullptr, out);
}
